// VectorQuantizer_54125177864310
// MI455X (gfx1250) — matrix operand order repaired and checked by reading, not compiled, not run on hardware
//
#include <hip/hip_runtime.h>
#include <hip/hip_bf16.h>

typedef __attribute__((ext_vector_type(16))) __bf16        v16bf;
typedef __attribute__((ext_vector_type(8)))  float          v8f;
typedef __attribute__((ext_vector_type(4)))  unsigned int   v4u;
typedef __attribute__((ext_vector_type(4)))  int            v4i;

#define B_    16
#define C_    256
#define H_    64
#define W_    64
#define NE_   1024
#define EDIM_ 256
#define NTOK_ (B_*H_*W_)            // 65536
#define LOSS_OFF (B_*C_*H_*W_)      // 16777216
#define CLS_OFF  (LOSS_OFF + 1)
#define IDX_OFF  (LOSS_OFF + 2)
#define TOTAL_ELEMS 16777216.0f     // B*C*H*W

// LDS staging: 32 codebook rows per tile, 256 bf16 each, padded to 264 (528B)
// so 16-lane fragment reads hit distinct banks (528B = 132 dw, stride 4 banks).
#define CB_ROWPAD 264

static __device__ __forceinline__ unsigned short f32_to_bf16_rne(float f) {
    unsigned int u = __float_as_uint(f);
    unsigned int r = u + 0x7FFFu + ((u >> 16) & 1u);
    return (unsigned short)(r >> 16);
}

// 16-byte async copy global -> LDS (CDNA5 GLOBAL_LOAD_ASYNC_TO_LDS_B128).
static __device__ __forceinline__ void async_cp16(void* lds, const void* g) {
#if __has_builtin(__builtin_amdgcn_global_load_async_to_lds_b128)
    __builtin_amdgcn_global_load_async_to_lds_b128(
        (v4i*)g, (v4i*)lds, 0, 0);
#else
    asm volatile("global_load_async_to_lds_b128 %0, %1, off"
                 :: "v"((unsigned int)(unsigned long long)lds), "v"(g) : "memory");
#endif
}

template <int N>
static __device__ __forceinline__ void s_wait_async() {
#if __has_builtin(__builtin_amdgcn_s_wait_asynccnt)
    __builtin_amdgcn_s_wait_asynccnt(N);
#else
    asm volatile("s_wait_asynccnt %0" :: "i"(N));
#endif
}

// ---------------------------------------------------------------------------
// Prep: codebook fp32 -> bf16 (row-major == column-major B), row norms, zero
// the two loss scalars in d_out.
// ---------------------------------------------------------------------------
__global__ __launch_bounds__(256) void vq_prep_kernel(
        const float* __restrict__ cb,
        unsigned short* __restrict__ cb16,
        float* __restrict__ cbn,
        float* __restrict__ out) {
    __shared__ float s[256];
    int row = blockIdx.x;          // 0..1023
    int t   = threadIdx.x;         // 0..255
    float v = cb[(size_t)row * EDIM_ + t];
    cb16[(size_t)row * EDIM_ + t] = f32_to_bf16_rne(v);
    s[t] = v * v;
    __syncthreads();
    #pragma unroll
    for (int off = 128; off > 0; off >>= 1) {
        if (t < off) s[t] += s[t + off];
        __syncthreads();
    }
    if (t == 0) cbn[row] = s[0];
    if (row == 0 && t < 2) out[LOSS_OFF + t] = 0.0f;   // codebook_loss, cls_loss
}

// ---------------------------------------------------------------------------
// Main: 256 threads = 8 waves, 16 tokens/wave = 128 tokens/block.
// Codebook tiles (32 codes) staged global->LDS with double-buffered async DMA,
// consumed by all 8 waves; two N-subtiles per iteration for WMMA ILP.
// ---------------------------------------------------------------------------
__global__ __launch_bounds__(256) void vq_main_kernel(
        const float* __restrict__ z,
        const float* __restrict__ cb,
        const unsigned short* __restrict__ cb16,
        const float* __restrict__ cbn,
        float* __restrict__ out) {
    __shared__ float s_cbn[NE_];
    __shared__ int   s_idx[128];
    __shared__ __align__(16) unsigned short s_cbt[2][32 * CB_ROWPAD];

    int tid = threadIdx.x;
    for (int i = tid; i < NE_; i += 256) s_cbn[i] = cbn[i];

    int wave = tid >> 5;
    int lane = tid & 31;
    int half = lane >> 4;          // 0: K-base 0 / M 0-7,  1: K-base 8 / M 8-15
    int col  = lane & 15;          // B/C column; also A row (token within tile)

    int tile = blockIdx.x * 8 + wave;      // 4096 tiles of 16 tokens
    int n0   = tile * 16;
    int w0   = n0 & 63;
    int h    = (n0 >> 6) & 63;
    int b    = n0 >> 12;

    // z[b][c][h][w] flat = ((b*C_ + c)*H_ + h)*W_ + w ; per-c stride 4096
    const float* zbase = z + ((size_t)b * C_ * H_ + h) * W_ + w0;
    const float* za    = zbase + col;

    union V16 { v16bf bf; unsigned short u[16]; v4u q[2]; };

    // Build all 8 A fragments (16x32 bf16), K = 0..255, in registers.
    V16 A[8];
    #pragma unroll
    for (int ks = 0; ks < 8; ++ks) {
        int kb = ks * 32 + half * 8;
        #pragma unroll
        for (int j = 0; j < 8; ++j) {
            A[ks].u[j]     = f32_to_bf16_rne(za[(size_t)(kb + j)      * 4096]);
            A[ks].u[j + 8] = f32_to_bf16_rne(za[(size_t)(kb + 16 + j) * 4096]);
        }
    }

    // Async stage of codebook tile nt (32 rows x 512B): thread t copies 4x16B,
    // contiguous 64B from global, into padded LDS rows.
    int st_row   = tid >> 3;          // 0..31
    int st_chunk = (tid & 7) * 4;     // 16B chunks within row
    auto issue_tile = [&](int nt, int buf) {
        const unsigned short* g =
            cb16 + ((size_t)(nt * 32 + st_row) * EDIM_ + st_chunk * 8);
        unsigned short* l = &s_cbt[buf][st_row * CB_ROWPAD + st_chunk * 8];
        #pragma unroll
        for (int c = 0; c < 4; ++c)
            async_cp16(l + c * 8, g + c * 8);
    };

    float minv[8];
    int   mini[8];
    #pragma unroll
    for (int r = 0; r < 8; ++r) { minv[r] = 3.4e38f; mini[r] = 0; }

    issue_tile(0, 0);

    #pragma unroll 1
    for (int nt = 0; nt < 32; ++nt) {
        int p = nt & 1;
        if (nt + 1 < 32) {
            issue_tile(nt + 1, p ^ 1);
            s_wait_async<4>();        // 4 in flight (next tile); current done
        } else {
            s_wait_async<0>();
        }
        __syncthreads();              // tile p visible to all 8 waves

        const v4u* b0 = (const v4u*)&s_cbt[p][(col)      * CB_ROWPAD];
        const v4u* b1 = (const v4u*)&s_cbt[p][(16 + col) * CB_ROWPAD];
        v8f acc0 = {}, acc1 = {};
        #pragma unroll
        for (int ks = 0; ks < 8; ++ks) {
            V16 B0, B1;
            B0.q[0] = b0[ks * 4 + half];
            B0.q[1] = b0[ks * 4 + half + 2];
            B1.q[0] = b1[ks * 4 + half];
            B1.q[1] = b1[ks * 4 + half + 2];
            acc0 = __builtin_amdgcn_wmma_f32_16x16x32_bf16(
                       false, A[ks].bf, false, B0.bf, (short)0, acc0, false, false);
            acc1 = __builtin_amdgcn_wmma_f32_16x16x32_bf16(
                       false, A[ks].bf, false, B1.bf, (short)0, acc1, false, false);
        }
        int   nA  = nt * 32 + col;
        int   nB  = nA + 16;
        float cn0 = s_cbn[nA];
        float cn1 = s_cbn[nB];
        #pragma unroll
        for (int r = 0; r < 8; ++r) {
            float d0 = cn0 - 2.0f * acc0[r];   // ||z||^2 constant: dropped
            if (d0 < minv[r]) { minv[r] = d0; mini[r] = nA; }
            float d1 = cn1 - 2.0f * acc1[r];
            if (d1 < minv[r]) { minv[r] = d1; mini[r] = nB; }
        }
        __syncthreads();              // all waves done with tile p before reuse
    }

    // Reduce across the 16 lanes of each half (xor masks stay within half).
    #pragma unroll
    for (int r = 0; r < 8; ++r) {
        float mv = minv[r]; int mi = mini[r];
        #pragma unroll
        for (int off = 1; off < 16; off <<= 1) {
            float ov = __shfl_xor(mv, off, 32);
            int   oi = __shfl_xor(mi, off, 32);
            if (ov < mv || (ov == mv && oi < mi)) { mv = ov; mi = oi; }
        }
        minv[r] = mv; mini[r] = mi;
    }

    if (col == 0) {
        #pragma unroll
        for (int r = 0; r < 8; ++r) {
            int m = half * 8 + r;
            s_idx[wave * 16 + m] = mini[r];
            out[IDX_OFF + n0 + m] = (float)mini[r];
        }
    }
    __syncthreads();

    // Gather + straight-through output (numerically z_q) + loss accumulation.
    int m   = col;
    int idx = s_idx[wave * 16 + m];
    float lsum = 0.0f;
    #pragma unroll 1
    for (int c0 = 0; c0 < C_; c0 += 2) {
        int c = c0 + half;
        float cv = cb[(size_t)idx * EDIM_ + c];
        size_t zo = ((size_t)(b * C_ + c) * H_ + h) * W_ + w0 + m;
        float zv = z[zo];
        out[zo] = cv;                       // z + sg(z_q - z) == z_q
        float diff = cv - zv;
        lsum += diff * diff;
    }
    #pragma unroll
    for (int off = 1; off < 32; off <<= 1) lsum += __shfl_xor(lsum, off, 32);
    if (lane == 0)
        atomicAdd(out + LOSS_OFF, lsum * (2.0f / TOTAL_ELEMS)); // q_loss + e_loss
}

extern "C" void kernel_launch(void* const* d_in, const int* in_sizes, int n_in,
                              void* d_out, int out_size, void* d_ws, size_t ws_size,
                              hipStream_t stream) {
    const float* z  = (const float*)d_in[0];
    const float* cb = (const float*)d_in[1];

    unsigned short* cb16 = (unsigned short*)d_ws;                       // 512 KB
    float* cbn = (float*)((char*)d_ws + (size_t)NE_ * EDIM_ * 2);       //   4 KB
    float* out = (float*)d_out;

    vq_prep_kernel<<<NE_, EDIM_, 0, stream>>>(cb, cb16, cbn, out);
    vq_main_kernel<<<NTOK_ / 128, 256, 0, stream>>>(z, cb, cb16, cbn, out);
}
